// EncoderDecoder_80461917323315
// MI455X (gfx1250) — matrix operand order repaired and checked by reading, not compiled, not run on hardware
//
#include <hip/hip_runtime.h>
#include <math.h>

#define BDIM 8192
#define FPD  4860
#define FPP  4864   // K padded to multiple of 32
#define ED   256
#define HD   256
#define VD   42
#define VP   48     // padded vocab
#define TT   128

typedef __attribute__((ext_vector_type(16))) __bf16 v16bf;
typedef __attribute__((ext_vector_type(8)))  __bf16 v8bf;
typedef __attribute__((ext_vector_type(8)))  float  v8f;

__device__ __forceinline__ unsigned short f2bf(float f) {
  unsigned int u = __builtin_bit_cast(unsigned int, f);
  u += 0x7fffu + ((u >> 16) & 1u);            // round-to-nearest-even
  return (unsigned short)(u >> 16);
}

// A fragment (16x32 bf16, row-major source): lane l<16 -> row m0+l, K
// kk..kk+15 (contiguous); lane l>=16 -> row m0+(l-16), K kk+16..kk+31.
__device__ __forceinline__ v16bf load_a_frag(const unsigned short* A, long pitch,
                                             int m0, int kk, int lane) {
  const __bf16* base = (const __bf16*)A;
  int r  = m0 + (lane & 15);
  int kb = kk + ((lane >> 4) << 4);
  const __bf16* p = base + (long)r * pitch + kb;
  v8bf lo = *(const v8bf*)p;
  v8bf hi = *(const v8bf*)(p + 8);
  return __builtin_shufflevector(lo, hi, 0,1,2,3,4,5,6,7,8,9,10,11,12,13,14,15);
}

// B fragment (32x16 bf16) sourced from row-major W (N x K): B[k][n] = W[n][k].
// lane l<16 -> col n0+l, K kk..kk+15 (contiguous); lane l>=16 -> K kk+16..kk+31.
__device__ __forceinline__ v16bf load_b_frag(const unsigned short* W, long pitch,
                                             int n0, int kk, int lane) {
  const __bf16* base = (const __bf16*)W;
  int n  = n0 + (lane & 15);
  int kb = kk + ((lane >> 4) << 4);
  return *(const v16bf*)(base + (long)n * pitch + kb);
}

__device__ __forceinline__ v8f wmma_bf16(v16bf a, v16bf b, v8f c) {
  return __builtin_amdgcn_wmma_f32_16x16x32_bf16(false, a, false, b, (short)0, c,
                                                 false, false);
}

__device__ __forceinline__ float sigmoidf_(float x) {
  return 1.f / (1.f + __expf(-x));
}

// ---------------------------------------------------------------- cast / fill
__global__ void cast_pad_kernel(unsigned short* __restrict__ dst,
                                const float* __restrict__ src,
                                int srows, int scols, int dcols, long total) {
  long i = blockIdx.x * (long)blockDim.x + threadIdx.x;
  if (i >= total) return;
  long r = i / dcols;
  int  c = (int)(i - r * dcols);
  float v = (r < srows && c < scols) ? src[r * (long)scols + c] : 0.f;
  dst[i] = f2bf(v);
}

__global__ void fill_zero_kernel(unsigned int* __restrict__ p, long n) {
  long i = blockIdx.x * (long)blockDim.x + threadIdx.x;
  if (i < n) p[i] = 0u;
}

// ------------------------------------------------- encoder GEMM + bias + ReLU
// C(M x N) = relu(A(M x K) @ W(N x K)^T + b), bf16 in/out, fp32 accumulate.
// Block: 256 thr = 8 waves (2 M-groups x 4 N-groups); wave tile 32x64:
// 8 WMMAs per K-step against 2 A-frags + 4 B-frags (12 x b128 loads).
__global__ void __launch_bounds__(256)
gemm_bias_relu_bf16(const unsigned short* __restrict__ A, long pA,
                    const unsigned short* __restrict__ W, long pW,
                    const float* __restrict__ bias,
                    unsigned short* __restrict__ out, long pOut, int K) {
  int lane = threadIdx.x & 31;
  int wave = threadIdx.x >> 5;
  int m0 = blockIdx.x * 64  + (wave & 1) * 32;
  int n0 = blockIdx.y * 256 + (wave >> 1) * 64;
  v8f c[2][4] = {};
  for (int kk = 0; kk < K; kk += 32) {
    if (kk + 128 < K)  // keep the weight stream warm (global_prefetch_b8)
      __builtin_prefetch((const __bf16*)W + (long)(n0 + (lane & 15)) * pW + kk + 128, 0, 1);
    v16bf a0 = load_a_frag(A, pA, m0,      kk, lane);
    v16bf a1 = load_a_frag(A, pA, m0 + 16, kk, lane);
    v16bf b[4];
#pragma unroll
    for (int j = 0; j < 4; ++j) b[j] = load_b_frag(W, pW, n0 + 16 * j, kk, lane);
#pragma unroll
    for (int j = 0; j < 4; ++j) {
      c[0][j] = wmma_bf16(a0, b[j], c[0][j]);
      c[1][j] = wmma_bf16(a1, b[j], c[1][j]);
    }
  }
  int half = lane >> 4, cl = lane & 15;
#pragma unroll
  for (int j = 0; j < 4; ++j) {
    int col = n0 + 16 * j + cl;
    float bv = bias[col];
#pragma unroll
    for (int i = 0; i < 2; ++i) {
#pragma unroll
      for (int r = 0; r < 8; ++r) {
        int row = m0 + 16 * i + r + half * 8;
        float v = c[i][j][r] + bv;
        out[(long)row * pOut + col] = f2bf(v > 0.f ? v : 0.f);
      }
    }
  }
}

// -------------------------------------------------------------- GRU cell step
// Block: 256 thr = 8 waves; wave owns 32 rows x 16 hidden cols: the six gate
// weight fragments are reused across two row tiles -> 12 WMMAs per K-step.
__global__ void __launch_bounds__(256)
gru_step_kernel(const unsigned short* __restrict__ xt,   // B x 256 bf16
                const unsigned short* __restrict__ hbf,  // B x 256 bf16 (prev)
                const float* __restrict__ hf,            // B x 256 f32  (prev)
                const unsigned short* __restrict__ wih,  // 768 x 256 bf16
                const unsigned short* __restrict__ whh,  // 768 x 256 bf16
                const float* __restrict__ bih, const float* __restrict__ bhh,
                float* __restrict__ hf_out, unsigned short* __restrict__ hbf_out) {
  int lane = threadIdx.x & 31;
  int wave = threadIdx.x >> 5;
  int m0 = blockIdx.x * 256 + wave * 32;
  int j0 = blockIdx.y * 16;
  v8f gi[3][2] = {};   // [gate r/z/n][row tile]
  v8f gh[3][2] = {};
  for (int kk = 0; kk < HD; kk += 32) {
    v16bf ax0 = load_a_frag(xt,  HD, m0,      kk, lane);
    v16bf ax1 = load_a_frag(xt,  HD, m0 + 16, kk, lane);
    v16bf ah0 = load_a_frag(hbf, HD, m0,      kk, lane);
    v16bf ah1 = load_a_frag(hbf, HD, m0 + 16, kk, lane);
    v16bf bi[3], bh[3];
#pragma unroll
    for (int g = 0; g < 3; ++g) {
      bi[g] = load_b_frag(wih, HD, j0 + 256 * g, kk, lane);
      bh[g] = load_b_frag(whh, HD, j0 + 256 * g, kk, lane);
    }
#pragma unroll
    for (int g = 0; g < 3; ++g) {
      gi[g][0] = wmma_bf16(ax0, bi[g], gi[g][0]);
      gi[g][1] = wmma_bf16(ax1, bi[g], gi[g][1]);
      gh[g][0] = wmma_bf16(ah0, bh[g], gh[g][0]);
      gh[g][1] = wmma_bf16(ah1, bh[g], gh[g][1]);
    }
  }
  int half = lane >> 4, cl = lane & 15;
  int col = j0 + cl;
  float br_i = bih[col], bz_i = bih[col + 256], bn_i = bih[col + 512];
  float br_h = bhh[col], bz_h = bhh[col + 256], bn_h = bhh[col + 512];
#pragma unroll
  for (int i = 0; i < 2; ++i) {
#pragma unroll
    for (int r = 0; r < 8; ++r) {
      int row = m0 + 16 * i + r + half * 8;
      float rg = sigmoidf_(gi[0][i][r] + br_i + gh[0][i][r] + br_h);
      float zg = sigmoidf_(gi[1][i][r] + bz_i + gh[1][i][r] + bz_h);
      float ng = tanhf(gi[2][i][r] + bn_i + rg * (gh[2][i][r] + bn_h));
      float ho = hf[(long)row * HD + col];
      float hn = (1.f - zg) * ng + zg * ho;
      hf_out[(long)row * HD + col]  = hn;
      hbf_out[(long)row * HD + col] = f2bf(hn);
    }
  }
}

// ----------------------------------------------- fc (H->42) + ReLU + softmax
// Block: 128 thr = 4 waves; wave owns 16 rows x (3 x 16) padded vocab tiles.
__global__ void __launch_bounds__(128)
fc_softmax_kernel(const unsigned short* __restrict__ h1,
                  const unsigned short* __restrict__ wfc,  // 48 x 256 bf16 (padded)
                  const float* __restrict__ bfc,
                  float* __restrict__ out, int t) {
  __shared__ float lds[64 * VP];
  int lane = threadIdx.x & 31;
  int wave = threadIdx.x >> 5;
  int m0 = blockIdx.x * 64 + wave * 16;
  v8f c0 = {}, c1 = {}, c2 = {};
  for (int kk = 0; kk < HD; kk += 32) {
    v16bf a  = load_a_frag(h1,  HD, m0, kk, lane);
    v16bf b0 = load_b_frag(wfc, HD, 0,  kk, lane);
    v16bf b1 = load_b_frag(wfc, HD, 16, kk, lane);
    v16bf b2 = load_b_frag(wfc, HD, 32, kk, lane);
    c0 = wmma_bf16(a, b0, c0);
    c1 = wmma_bf16(a, b1, c1);
    c2 = wmma_bf16(a, b2, c2);
  }
  int half = lane >> 4, cl = lane & 15;
  float bb0 = bfc[cl];
  float bb1 = bfc[16 + cl];
  float bb2 = (32 + cl) < VD ? bfc[32 + cl] : 0.f;
#pragma unroll
  for (int r = 0; r < 8; ++r) {
    int rl = wave * 16 + r + half * 8;
    float v0 = c0[r] + bb0, v1 = c1[r] + bb1, v2 = c2[r] + bb2;
    lds[rl * VP + cl]      = v0 > 0.f ? v0 : 0.f;
    lds[rl * VP + 16 + cl] = v1 > 0.f ? v1 : 0.f;
    lds[rl * VP + 32 + cl] = v2 > 0.f ? v2 : 0.f;
  }
  __syncthreads();
  if (threadIdx.x < 64) {
    int rl = threadIdx.x;
    long b = (long)blockIdx.x * 64 + rl;
    float* rowp = &lds[rl * VP];
    float mx = rowp[0];
    for (int v = 1; v < VD; ++v) mx = fmaxf(mx, rowp[v]);
    float s = 0.f;
    for (int v = 0; v < VD; ++v) { float e = __expf(rowp[v] - mx); rowp[v] = e; s += e; }
    float inv = 1.f / s;
    float* o = out + (b * TT + t) * (long)VD;
    for (int v = 0; v < VD; ++v) o[v] = rowp[v] * inv;
  }
}

// ----------------------------------------------------------------------- host
extern "C" void kernel_launch(void* const* d_in, const int* in_sizes, int n_in,
                              void* d_out, int out_size, void* d_ws, size_t ws_size,
                              hipStream_t stream) {
  (void)in_sizes; (void)n_in; (void)out_size;
  const float* x    = (const float*)d_in[0];
  const float* w1   = (const float*)d_in[1];
  const float* b1   = (const float*)d_in[2];
  const float* w2   = (const float*)d_in[3];
  const float* b2   = (const float*)d_in[4];
  const float* w3   = (const float*)d_in[5];
  const float* b3   = (const float*)d_in[6];
  const float* w4   = (const float*)d_in[7];
  const float* b4   = (const float*)d_in[8];
  const float* wih0 = (const float*)d_in[9];
  const float* whh0 = (const float*)d_in[10];
  const float* bih0 = (const float*)d_in[11];
  const float* bhh0 = (const float*)d_in[12];
  const float* wih1 = (const float*)d_in[13];
  const float* whh1 = (const float*)d_in[14];
  const float* bih1 = (const float*)d_in[15];
  const float* bhh1 = (const float*)d_in[16];
  const float* wfc  = (const float*)d_in[17];
  const float* bfc  = (const float*)d_in[18];

  char* ws = (char*)d_ws;
  size_t off = 0;
  auto take = [&](size_t bytes) -> char* {
    char* p = ws + off;
    off = (off + bytes + 255) & ~(size_t)255;
    return p;
  };
  unsigned short* x_bf    = (unsigned short*)take((size_t)BDIM * FPP * 2);
  unsigned short* w1_bf   = (unsigned short*)take((size_t)2048 * FPP * 2);
  unsigned short* a1_bf   = (unsigned short*)take((size_t)BDIM * 2048 * 2);
  unsigned short* w2_bf   = (unsigned short*)take((size_t)1024 * 2048 * 2);
  unsigned short* w3_bf   = (unsigned short*)take((size_t)512 * 1024 * 2);
  unsigned short* w4_bf   = (unsigned short*)take((size_t)256 * 512 * 2);
  unsigned short* e_bf    = (unsigned short*)take((size_t)BDIM * ED * 2);
  unsigned short* wih0_bf = (unsigned short*)take((size_t)768 * 256 * 2);
  unsigned short* whh0_bf = (unsigned short*)take((size_t)768 * 256 * 2);
  unsigned short* wih1_bf = (unsigned short*)take((size_t)768 * 256 * 2);
  unsigned short* whh1_bf = (unsigned short*)take((size_t)768 * 256 * 2);
  unsigned short* wfc_bf  = (unsigned short*)take((size_t)VP * 256 * 2);
  float*          h0f[2]; float* h1f[2];
  unsigned short* h0b[2]; unsigned short* h1b[2];
  for (int i = 0; i < 2; ++i) h0f[i] = (float*)take((size_t)BDIM * HD * 4);
  for (int i = 0; i < 2; ++i) h1f[i] = (float*)take((size_t)BDIM * HD * 4);
  for (int i = 0; i < 2; ++i) h0b[i] = (unsigned short*)take((size_t)BDIM * HD * 2);
  for (int i = 0; i < 2; ++i) h1b[i] = (unsigned short*)take((size_t)BDIM * HD * 2);
  // activation buffer reuse: a2 lives in the (larger) x_bf region, a3 in a1's
  unsigned short* a2_bf = x_bf;
  unsigned short* a3_bf = a1_bf;
  if (off > ws_size) return;  // workspace too small; nothing safe to do

  auto cast = [&](unsigned short* dst, const float* src,
                  int srows, int scols, int drows, int dcols) {
    long total = (long)drows * dcols;
    unsigned int nb = (unsigned int)((total + 255) / 256);
    cast_pad_kernel<<<dim3(nb), dim3(256), 0, stream>>>(dst, src, srows, scols, dcols, total);
  };
  auto zero = [&](void* p, long bytes) {
    long n = bytes / 4;
    fill_zero_kernel<<<dim3((unsigned)((n + 255) / 256)), dim3(256), 0, stream>>>(
        (unsigned int*)p, n);
  };

  // fp32 -> bf16 (with K-padding for layer 1, row-padding for wfc)
  cast(x_bf,    x,    BDIM, FPD, BDIM, FPP);
  cast(w1_bf,   w1,   2048, FPD, 2048, FPP);
  cast(w2_bf,   w2,   1024, 2048, 1024, 2048);
  cast(w3_bf,   w3,   512, 1024, 512, 1024);
  cast(w4_bf,   w4,   256, 512, 256, 512);
  cast(wih0_bf, wih0, 768, 256, 768, 256);
  cast(whh0_bf, whh0, 768, 256, 768, 256);
  cast(wih1_bf, wih1, 768, 256, 768, 256);
  cast(whh1_bf, whh1, 768, 256, 768, 256);
  cast(wfc_bf,  wfc,  VD, 256, VP, 256);

  // encoder MLP (block tile 64 x 256)
  gemm_bias_relu_bf16<<<dim3(BDIM / 64, 2048 / 256), dim3(256), 0, stream>>>(
      x_bf, FPP, w1_bf, FPP, b1, a1_bf, 2048, FPP);
  gemm_bias_relu_bf16<<<dim3(BDIM / 64, 1024 / 256), dim3(256), 0, stream>>>(
      a1_bf, 2048, w2_bf, 2048, b2, a2_bf, 1024, 2048);
  gemm_bias_relu_bf16<<<dim3(BDIM / 64, 512 / 256), dim3(256), 0, stream>>>(
      a2_bf, 1024, w3_bf, 1024, b3, a3_bf, 512, 1024);
  gemm_bias_relu_bf16<<<dim3(BDIM / 64, 256 / 256), dim3(256), 0, stream>>>(
      a3_bf, 512, w4_bf, 512, b4, e_bf, 256, 512);

  // zero initial GRU states (ping index 0)
  zero(h0f[0], (long)BDIM * HD * 4);
  zero(h1f[0], (long)BDIM * HD * 4);
  zero(h0b[0], (long)BDIM * HD * 2);
  zero(h1b[0], (long)BDIM * HD * 2);

  // autoregressive GRU stack + fc/softmax, ping-pong states across steps
  for (int t = 0; t < TT; ++t) {
    int p = t & 1, q = p ^ 1;
    const unsigned short* xt = (t == 0) ? e_bf : h1b[p];
    gru_step_kernel<<<dim3(BDIM / 256, HD / 16), dim3(256), 0, stream>>>(
        xt, h0b[p], h0f[p], wih0_bf, whh0_bf, bih0, bhh0, h0f[q], h0b[q]);
    gru_step_kernel<<<dim3(BDIM / 256, HD / 16), dim3(256), 0, stream>>>(
        h0b[q], h1b[p], h1f[p], wih1_bf, whh1_bf, bih1, bhh1, h1f[q], h1b[q]);
    fc_softmax_kernel<<<dim3(BDIM / 64), dim3(128), 0, stream>>>(
        h1b[q], wfc_bf, bfc, (float*)d_out, t);
  }
}
